// PCERegressor_59279138620021
// MI455X (gfx1250) — matrix operand order repaired and checked by reading, not compiled, not run on hardware
//
#include <hip/hip_runtime.h>
#include <hip/hip_bf16.h>

typedef _Float16 v16h __attribute__((ext_vector_type(16)));
typedef float    v8f  __attribute__((ext_vector_type(8)));

__host__ __device__ constexpr int ilog2c(int v) { return (v <= 1) ? 0 : 1 + ilog2c(v >> 1); }

// ---------------------------------------------------------------------------
// Edge MLP stage 1: H[e,:] = relu(edge_attr[e,:16] @ Wa[16,HID] + ba)
// one block per edge, blockDim = HID
// ---------------------------------------------------------------------------
template <int HID>
__global__ void k_edge_mlp(const float* __restrict__ EA, const float* __restrict__ Wa,
                           const float* __restrict__ ba, float* __restrict__ H) {
  __shared__ float ea[16];
  const int e = blockIdx.x, k = threadIdx.x;
  if (k < 16) ea[k] = EA[e * 16 + k];
  __syncthreads();
  float s = ba[k];
#pragma unroll
  for (int f = 0; f < 16; ++f) s = fmaf(ea[f], Wa[f * HID + k], s);
  H[(size_t)e * HID + k] = fmaxf(s, 0.f);
}

// ---------------------------------------------------------------------------
// Repack [Wb ; bb] (f32) into f16 WMMA B-fragment layout.
// Logical B': [(HID+1)*IN, OUT], row j = k*IN + i maps to Wb_flat[j*OUT+o]
// (bias rows j >= HID*IN map to bb[(j-HID*IN)*OUT+o]).
// Layout this code uses (contiguous K, same as A; 32x16 KxN): lanes 0-15 hold K=0..15 of column
// n=lane, lanes 16-31 hold K=16..31 of column n=lane-16; element c = K%16.
// Storage: BF[(((kc*nNt + nt)<<5) + lane)*16 + c]
// ---------------------------------------------------------------------------
template <int HID, int IN, int OUT>
__global__ void k_repack_b(const float* __restrict__ Wb, const float* __restrict__ bb,
                           _Float16* __restrict__ BF) {
  constexpr int KT  = (HID + 1) * IN;
  constexpr int nNt = OUT / 16;
  const int idx = blockIdx.x * blockDim.x + threadIdx.x;
  if (idx >= KT * OUT) return;
  const int j = idx / OUT, o = idx % OUT;
  const float v = (j < HID * IN) ? Wb[(size_t)j * OUT + o] : bb[(size_t)(j - HID * IN) * OUT + o];
  const int kc = j >> 5, kk = j & 31, nt = o >> 4, n = o & 15;
  const int lane = n + ((kk >> 4) << 4);
  const int c = kk & 15;
  BF[(((size_t)(kc * nNt + nt) << 5) + lane) * 16 + c] = (_Float16)v;
}

// ---------------------------------------------------------------------------
// Fused message GEMM:  msg[e,o] = sum_j A[e,j]*B'[j,o],
//   A[e,j] = H[e, j>>log2(IN)] * x[src[e], j&(IN-1)]   (H row HID := 1.0 for bias)
// One wave per (16-edge tile, NGROUP N-tiles); f16 WMMA, f32 accumulate.
// Atomic-add into AGG[dst[e]*OUT + o].
// ---------------------------------------------------------------------------
template <int HID, int IN, int OUT, int NGROUP>
__global__ __launch_bounds__(32) void k_msg_wmma(
    const float* __restrict__ X, const float* __restrict__ H, const _Float16* __restrict__ BF,
    const int* __restrict__ src, const int* __restrict__ dst, float* __restrict__ AGG) {
  constexpr int KT    = (HID + 1) * IN;
  constexpr int nNt   = OUT / 16;
  constexpr int LOGIN = ilog2c(IN);
  static_assert((KT & 31) == 0, "K must be multiple of 32");

  __shared__ float Hs[16][HID + 4];
  __shared__ float Xs[16][IN];
  __shared__ int   Ss[16], Sd[16];

  const int lane = threadIdx.x;
  const int e0   = blockIdx.x * 16;

  if (lane < 16) {
    Ss[lane] = src[e0 + lane];
    Sd[lane] = dst[e0 + lane];
  }
  __syncthreads();
  // gather x[src] rows
  for (int t = lane; t < 16 * IN; t += 32) {
    const int r = t >> LOGIN, c = t & (IN - 1);
    Xs[r][c] = X[(size_t)Ss[r] * IN + c];
  }
  // stage H rows
  constexpr int LOGH = ilog2c(HID);
  for (int t = lane; t < 16 * HID; t += 32) {
    const int r = t >> LOGH, c = t & (HID - 1);
    Hs[r][c] = H[(size_t)(e0 + r) * HID + c];
  }
  if (lane < 16) Hs[lane][HID] = 1.0f;  // bias fold row
  __syncthreads();

  const int m      = lane & 15;
  const int base   = (lane >> 4) << 4;   // 0 or 16 (A fragment K split)
  const int ntBase = blockIdx.y * NGROUP;

  v8f acc[NGROUP];
#pragma unroll
  for (int g = 0; g < NGROUP; ++g)
#pragma unroll
    for (int r = 0; r < 8; ++r) acc[g][r] = 0.f;

  for (int kc = 0; kc < KT / 32; ++kc) {
    const int j0 = kc << 5;
    v16h a;
    {
      const int jr = j0 + base;                       // 8-aligned, stays in one k
      const float  h0 = Hs[m][jr >> LOGIN];
      const float* xr = &Xs[m][jr & (IN - 1)];
#pragma unroll
      for (int t = 0; t < 8; ++t) a[t] = (_Float16)(h0 * xr[t]);
      const int jr2 = jr + 8;
      const float  h1  = Hs[m][jr2 >> LOGIN];
      const float* xr2 = &Xs[m][jr2 & (IN - 1)];
#pragma unroll
      for (int t = 0; t < 8; ++t) a[8 + t] = (_Float16)(h1 * xr2[t]);
    }
#pragma unroll
    for (int g = 0; g < NGROUP; ++g) {
      const int nt = ntBase + g;
      const v16h b = *reinterpret_cast<const v16h*>(
          BF + ((((size_t)kc * nNt + nt) << 5) + lane) * 16);
      acc[g] = __builtin_amdgcn_wmma_f32_16x16x32_f16(false, a, false, b, (short)0, acc[g],
                                                      false, false);
    }
  }

  // D layout: lane holds rows r + 8*(lane>=16), column lane&15
  const int hi = lane >> 4, n = lane & 15;
#pragma unroll
  for (int g = 0; g < NGROUP; ++g) {
    const int o = (ntBase + g) * 16 + n;
#pragma unroll
    for (int r = 0; r < 8; ++r) {
      const int row = r + (hi << 3);
      atomicAdd(&AGG[(size_t)Sd[row] * OUT + o], acc[g][r]);
    }
  }
}

// ---------------------------------------------------------------------------
// In-degree counts (mean aggregation denominators)
// ---------------------------------------------------------------------------
__global__ void k_count(const int* __restrict__ dst, float* __restrict__ CNT, int E) {
  const int e = blockIdx.x * blockDim.x + threadIdx.x;
  if (e < E) atomicAdd(&CNT[dst[e]], 1.f);
}

// ---------------------------------------------------------------------------
// Node update: t = agg/cnt + hprev@root + cb ; BN(eval) ; relu ; x*sigmoid(x@aw+ab)
// blockDim = 128; 128/OUT nodes per block
// ---------------------------------------------------------------------------
template <int INL, int OUT>
__global__ void k_node_update(const float* __restrict__ AGG, const float* __restrict__ CNT,
                              const float* __restrict__ hprev, const float* __restrict__ root,
                              const float* __restrict__ cb, const float* __restrict__ gn,
                              const float* __restrict__ be, const float* __restrict__ mm,
                              const float* __restrict__ vv, const float* __restrict__ aw,
                              const float* __restrict__ ab, float* __restrict__ hn, int N) {
  constexpr int NPB = 128 / OUT;
  __shared__ float rowbuf[NPB][INL];
  __shared__ float red[128];
  const int sub = threadIdx.x / OUT, o = threadIdx.x % OUT;
  const int node = blockIdx.x * NPB + sub;

  for (int i = o; i < INL; i += OUT)
    rowbuf[sub][i] = (node < N) ? hprev[(size_t)node * INL + i] : 0.f;
  __syncthreads();

  float t = 0.f;
  if (node < N) {
    float c = CNT[node];
    c = (c < 1.f) ? 1.f : c;
    t = AGG[(size_t)node * OUT + o] / c;
    float d = 0.f;
#pragma unroll 4
    for (int i = 0; i < INL; ++i) d = fmaf(rowbuf[sub][i], root[i * OUT + o], d);
    t += d + cb[o];
    t = (t - mm[o]) * rsqrtf(vv[o] + 1e-5f) * gn[o] + be[o];
    t = fmaxf(t, 0.f);
  }
  red[threadIdx.x] = t * aw[o];
  __syncthreads();
  for (int s = OUT / 2; s > 0; s >>= 1) {
    if (o < s) red[threadIdx.x] += red[threadIdx.x + s];
    __syncthreads();
  }
  const float gate = 1.f / (1.f + __expf(-(red[sub * OUT] + ab[0])));
  if (node < N) hn[(size_t)node * OUT + o] = t * gate;
}

// ---------------------------------------------------------------------------
// Global mean pool (sums + counts)
// ---------------------------------------------------------------------------
__global__ void k_pool(const float* __restrict__ h3, const int* __restrict__ batch,
                       float* __restrict__ GSUM, float* __restrict__ GCNT, int N) {
  const int idx = blockIdx.x * blockDim.x + threadIdx.x;
  if (idx >= N * 32) return;
  const int node = idx >> 5, o = idx & 31;
  const int g = batch[node];
  atomicAdd(&GSUM[(size_t)g * 32 + o], h3[idx]);
  if (o == 0) atomicAdd(&GCNT[g], 1.f);
}

// ---------------------------------------------------------------------------
// Head: g = mean pooled -> relu(fc1) -> fc2
// ---------------------------------------------------------------------------
__global__ void k_head(const float* __restrict__ GSUM, const float* __restrict__ GCNT,
                       const float* __restrict__ fc1w, const float* __restrict__ fc1b,
                       const float* __restrict__ fc2w, const float* __restrict__ fc2b,
                       float* __restrict__ out, int G) {
  const int g = blockIdx.x * blockDim.x + threadIdx.x;
  if (g >= G) return;
  float c = GCNT[g];
  c = (c < 1.f) ? 1.f : c;
  float gv[32];
#pragma unroll
  for (int i = 0; i < 32; ++i) gv[i] = GSUM[(size_t)g * 32 + i] / c;
  float acc = fc2b[0];
#pragma unroll
  for (int j = 0; j < 16; ++j) {
    float h = fc1b[j];
#pragma unroll
    for (int i = 0; i < 32; ++i) h = fmaf(gv[i], fc1w[i * 16 + j], h);
    acc = fmaf(fmaxf(h, 0.f), fc2w[j], acc);
  }
  out[g] = acc;
}

// ---------------------------------------------------------------------------
extern "C" void kernel_launch(void* const* d_in, const int* in_sizes, int n_in,
                              void* d_out, int out_size, void* d_ws, size_t ws_size,
                              hipStream_t stream) {
  const float* x   = (const float*)d_in[0];
  const int*   ei  = (const int*)d_in[1];
  const float* ea  = (const float*)d_in[2];
  const int*   bat = (const int*)d_in[3];
  const float *Wa1 = (const float*)d_in[5],  *ba1 = (const float*)d_in[6];
  const float *Wb1 = (const float*)d_in[7],  *bb1 = (const float*)d_in[8];
  const float *root1 = (const float*)d_in[9], *cb1 = (const float*)d_in[10];
  const float *g1 = (const float*)d_in[11], *be1 = (const float*)d_in[12];
  const float *m1 = (const float*)d_in[13], *v1  = (const float*)d_in[14];
  const float *aw1 = (const float*)d_in[15], *ab1 = (const float*)d_in[16];
  const float *Wa2 = (const float*)d_in[17], *ba2 = (const float*)d_in[18];
  const float *Wb2 = (const float*)d_in[19], *bb2 = (const float*)d_in[20];
  const float *root2 = (const float*)d_in[21], *cb2 = (const float*)d_in[22];
  const float *g2 = (const float*)d_in[23], *be2 = (const float*)d_in[24];
  const float *m2 = (const float*)d_in[25], *v2  = (const float*)d_in[26];
  const float *aw2 = (const float*)d_in[27], *ab2 = (const float*)d_in[28];
  const float *Wa3 = (const float*)d_in[29], *ba3 = (const float*)d_in[30];
  const float *Wb3 = (const float*)d_in[31], *bb3 = (const float*)d_in[32];
  const float *root3 = (const float*)d_in[33], *cb3 = (const float*)d_in[34];
  const float *g3 = (const float*)d_in[35], *be3 = (const float*)d_in[36];
  const float *m3 = (const float*)d_in[37], *v3  = (const float*)d_in[38];
  const float *aw3 = (const float*)d_in[39], *ab3 = (const float*)d_in[40];
  const float *fc1w = (const float*)d_in[41], *fc1b = (const float*)d_in[42];
  const float *fc2w = (const float*)d_in[43], *fc2b = (const float*)d_in[44];

  const int N = in_sizes[0] / 32;
  const int E = in_sizes[2] / 16;
  const int G = out_size;
  const int* src = ei;
  const int* dst = ei + E;

  // workspace carve-out (256B aligned)
  char* base = (char*)d_ws;
  size_t off = 0;
  auto carve = [&](size_t bytes) -> char* {
    char* p = base + off;
    off = (off + bytes + 255) & ~(size_t)255;
    return p;
  };
  float*    H    = (float*)carve((size_t)E * 128 * 4);      // edge hidden (reused)
  _Float16* BF   = (_Float16*)carve((size_t)16512 * 64 * 2); // max repacked B (layer2)
  float*    AGG  = (float*)carve((size_t)N * 128 * 4);       // message sums (reused)
  float*    CNT  = (float*)carve((size_t)N * 4);
  float*    HN1  = (float*)carve((size_t)N * 128 * 4);
  float*    HN2  = (float*)carve((size_t)N * 64 * 4);
  float*    HN3  = (float*)carve((size_t)N * 32 * 4);
  float*    GSUM = (float*)carve((size_t)G * 32 * 4);
  float*    GCNT = (float*)carve((size_t)G * 4);
  (void)ws_size; (void)n_in; (void)bat;

  const int etiles = E / 16;

  // in-degree counts (same graph for all layers)
  hipMemsetAsync(CNT, 0, (size_t)N * 4, stream);
  k_count<<<(E + 255) / 256, 256, 0, stream>>>(dst, CNT, E);

  // ---- Layer 1: IN=32, HID=128, OUT=128 ----
  hipMemsetAsync(AGG, 0, (size_t)N * 128 * 4, stream);
  k_edge_mlp<128><<<E, 128, 0, stream>>>(ea, Wa1, ba1, H);
  {
    constexpr int TOT = (128 + 1) * 32 * 128;
    k_repack_b<128, 32, 128><<<(TOT + 255) / 256, 256, 0, stream>>>(Wb1, bb1, BF);
  }
  k_msg_wmma<128, 32, 128, 4><<<dim3(etiles, 2), 32, 0, stream>>>(x, H, BF, src, dst, AGG);
  k_node_update<32, 128><<<N, 128, 0, stream>>>(AGG, CNT, x, root1, cb1, g1, be1, m1, v1,
                                                aw1, ab1, HN1, N);

  // ---- Layer 2: IN=128, HID=128, OUT=64 ----
  hipMemsetAsync(AGG, 0, (size_t)N * 64 * 4, stream);
  k_edge_mlp<128><<<E, 128, 0, stream>>>(ea, Wa2, ba2, H);
  {
    constexpr int TOT = (128 + 1) * 128 * 64;
    k_repack_b<128, 128, 64><<<(TOT + 255) / 256, 256, 0, stream>>>(Wb2, bb2, BF);
  }
  k_msg_wmma<128, 128, 64, 4><<<dim3(etiles, 1), 32, 0, stream>>>(HN1, H, BF, src, dst, AGG);
  k_node_update<128, 64><<<(N + 1) / 2, 128, 0, stream>>>(AGG, CNT, HN1, root2, cb2, g2, be2,
                                                          m2, v2, aw2, ab2, HN2, N);

  // ---- Layer 3: IN=64, HID=64, OUT=32 ----
  hipMemsetAsync(AGG, 0, (size_t)N * 32 * 4, stream);
  k_edge_mlp<64><<<E, 64, 0, stream>>>(ea, Wa3, ba3, H);
  {
    constexpr int TOT = (64 + 1) * 64 * 32;
    k_repack_b<64, 64, 32><<<(TOT + 255) / 256, 256, 0, stream>>>(Wb3, bb3, BF);
  }
  k_msg_wmma<64, 64, 32, 2><<<dim3(etiles, 1), 32, 0, stream>>>(HN2, H, BF, src, dst, AGG);
  k_node_update<64, 32><<<(N + 3) / 4, 128, 0, stream>>>(AGG, CNT, HN2, root3, cb3, g3, be3,
                                                         m3, v3, aw3, ab3, HN3, N);

  // ---- pool + head ----
  hipMemsetAsync(GSUM, 0, (size_t)G * 32 * 4, stream);
  hipMemsetAsync(GCNT, 0, (size_t)G * 4, stream);
  k_pool<<<(N * 32 + 255) / 256, 256, 0, stream>>>(HN3, bat, GSUM, GCNT, N);
  k_head<<<(G + 255) / 256, 256, 0, stream>>>(GSUM, GCNT, fc1w, fc1b, fc2w, fc2b,
                                              (float*)d_out, G);
}
